// GRNN_41532333752663
// MI455X (gfx1250) — matrix operand order repaired and checked by reading, not compiled, not run on hardware
//
#include <hip/hip_runtime.h>

// ---------------------------------------------------------------------------
// GRU (B=64, T=512, D=H=1024) for gfx1250 / MI455X, wave32 WMMA bf16 path.
//
// Phase 0 (prep):   Xt, W_ih, W_hh f32 -> bf16 in ws; h0 -> f32+bf16 ping
//                   buffers; reset grid-barrier counter.
// Phase 1 (gi):     Gi[B*T, 3H] = Xt @ W_ih^T + b_ih (WMMA GEMM, 128x64 tiles,
//                   bf16 operands preconverted -> no cvt in the inner loop)
// Phase 2 (recur):  persistent kernel, 128 blocks (4 batch-tiles x 32 col
//                   slices). W_hh slice + per-step h tile staged into LDS via
//                   global_load_async_to_lds_b128 (ASYNCcnt), WMMA from LDS,
//                   h double-buffered in global (L2-resident), device-wide
//                   monotone barrier per timestep.
// ---------------------------------------------------------------------------

#define Bsz 64
#define Tsz 512
#define Dsz 1024
#define Hsz 1024
#define H3  3072

typedef float v8f   __attribute__((ext_vector_type(8)));
typedef __bf16 v16bf __attribute__((ext_vector_type(16)));
typedef unsigned int u32x4 __attribute__((ext_vector_type(4)));

union Frag {
    v16bf v;
    u32x4 q[2];
    __bf16 e[16];
};

// ----------------------------- workspace layout ----------------------------
#define OFF_GI    ((size_t)0)                       // f32  [B*T,3H]  402,653,184
#define OFF_XTB   (OFF_GI   + (size_t)402653184)    // bf16 [B*T,D]    67,108,864
#define OFF_WIHB  (OFF_XTB  + (size_t)67108864)     // bf16 [3H,D]      6,291,456
#define OFF_WHHB  (OFF_WIHB + (size_t)6291456)      // bf16 [3H,H]      6,291,456
#define OFF_HB32  (OFF_WHHB + (size_t)6291456)      // f32  [2][B,H]      524,288
#define OFF_HB16  (OFF_HB32 + (size_t)524288)       // bf16 [2][B,H]      262,144
#define OFF_BAR   (OFF_HB16 + (size_t)262144)       // u32 barrier

// recurrent-kernel LDS (byte offsets into dynamic LDS, which starts at 0)
#define WPAD       1032
#define LDS_WL     0u                               // [96][WPAD] bf16  198,144 B
#define LDS_HT     (96u * WPAD * 2u)                // [16][WPAD] bf16   33,024 B
#define LDS_GHS    (LDS_HT + 16u * WPAD * 2u)       // [6][16][16] f32    6,144 B
#define SMEM_TOTAL (LDS_GHS + 6u * 256u * 4u)       // 237,312 B

// async copy helpers (gfx1250 async-to-LDS path, tracked by ASYNCcnt)
__device__ __forceinline__ void async_g2l_b128(unsigned lds_off, const void* gsrc) {
    asm volatile("global_load_async_to_lds_b128 %0, %1, off"
                 :: "v"(lds_off), "v"(gsrc) : "memory");
}
__device__ __forceinline__ void async_wait0() {
    asm volatile("s_wait_asynccnt 0x0" ::: "memory");
}

// ------------------------------- prep kernel -------------------------------
__global__ void grnn_prep(const float* __restrict__ Xt,
                          const float* __restrict__ Wih, const float* __restrict__ Whh,
                          const float* __restrict__ h0,
                          __bf16* __restrict__ XtB,
                          __bf16* __restrict__ WihB, __bf16* __restrict__ WhhB,
                          float* __restrict__ hb32, __bf16* __restrict__ hb16,
                          unsigned* __restrict__ bar) {
    size_t idx = (size_t)blockIdx.x * blockDim.x + threadIdx.x;
    size_t stride = (size_t)gridDim.x * blockDim.x;
    for (size_t i = idx; i < (size_t)Bsz * Tsz * Dsz; i += stride)
        XtB[i] = (__bf16)Xt[i];
    for (size_t i = idx; i < (size_t)H3 * Dsz; i += stride) {
        WihB[i] = (__bf16)Wih[i];
        WhhB[i] = (__bf16)Whh[i];
    }
    for (size_t i = idx; i < (size_t)Bsz * Hsz; i += stride) {
        float v = h0[i];
        hb32[i] = v;
        hb16[i] = (__bf16)v;
    }
    if (idx == 0) *bar = 0u;
}

// --------------------------- phase 1: Gi GEMM ------------------------------
// Gi[m, n] = sum_k XtB[m, k] * W_ih[n, k] + b_ih[n];  m = b*T + t.
// 256 threads (8 waves); block tile 128(M) x 64(N); wave -> 2 M-tiles x 2
// N-tiles (4 WMMAs / k-step with A/B fragment reuse).
__global__ void __launch_bounds__(256)
grnn_gi_gemm(const __bf16* __restrict__ X, const __bf16* __restrict__ Wb,
             const float* __restrict__ bih, float* __restrict__ Gi) {
    const int tid  = threadIdx.x;
    const int lane = tid & 31;
    const int wave = tid >> 5;
    const int ln   = lane & 15;
    const int lh   = lane >> 4;
    const int mp   = wave & 3;    // M pair: rows mp*32 .. +32
    const int np   = wave >> 2;   // N pair: cols np*32 .. +32

    const int Mbase = blockIdx.y * 128;
    const int Nbase = blockIdx.x * 64;

    v8f acc00 = {}, acc01 = {}, acc10 = {}, acc11 = {};
    const __bf16* xa0 = X  + (size_t)(Mbase + mp * 32 + ln) * Dsz;
    const __bf16* xa1 = xa0 + (size_t)16 * Dsz;
    const __bf16* w0  = Wb + (size_t)(Nbase + np * 32 + ln) * Dsz;
    const __bf16* w1  = w0 + (size_t)16 * Dsz;

    for (int ks = 0; ks < 32; ++ks) {
        Frag a0, a1, b0, b1;
        const int k0 = ks * 32 + lh * 8;   // A: K = k0..k0+7, k0+16..k0+23
        a0.q[0] = *(const u32x4*)(xa0 + k0);
        a0.q[1] = *(const u32x4*)(xa0 + k0 + 16);
        a1.q[0] = *(const u32x4*)(xa1 + k0);
        a1.q[1] = *(const u32x4*)(xa1 + k0 + 16);
        const int kb = ks * 32 + lh * 8;   // B: K = kb..kb+7, kb+16..kb+23
        b0.q[0] = *(const u32x4*)(w0 + kb);
        b0.q[1] = *(const u32x4*)(w0 + kb + 16);
        b1.q[0] = *(const u32x4*)(w1 + kb);
        b1.q[1] = *(const u32x4*)(w1 + kb + 16);
        acc00 = __builtin_amdgcn_wmma_f32_16x16x32_bf16(false, a0.v, false, b0.v,
                                                        (short)0, acc00, false, false);
        acc01 = __builtin_amdgcn_wmma_f32_16x16x32_bf16(false, a0.v, false, b1.v,
                                                        (short)0, acc01, false, false);
        acc10 = __builtin_amdgcn_wmma_f32_16x16x32_bf16(false, a1.v, false, b0.v,
                                                        (short)0, acc10, false, false);
        acc11 = __builtin_amdgcn_wmma_f32_16x16x32_bf16(false, a1.v, false, b1.v,
                                                        (short)0, acc11, false, false);
    }

    const int n0 = Nbase + np * 32 + ln;
    const int n1 = n0 + 16;
    const float bi0 = bih[n0], bi1 = bih[n1];
    #pragma unroll
    for (int r = 0; r < 8; ++r) {
        const int m0 = Mbase + mp * 32 + r + 8 * lh;       // C: M = r + 8*laneHi
        const int m1 = m0 + 16;
        Gi[(size_t)m0 * H3 + n0] = acc00[r] + bi0;
        Gi[(size_t)m0 * H3 + n1] = acc01[r] + bi1;
        Gi[(size_t)m1 * H3 + n0] = acc10[r] + bi0;
        Gi[(size_t)m1 * H3 + n1] = acc11[r] + bi1;
    }
}

// ------------------------- phase 2: recurrence -----------------------------
// grid = (32 column-slices, 4 batch-tiles); 192 threads = 6 waves.
// wave w: gate g = w>>1, sub-tile jj0 = (w&1)*16 of a 32-wide column slice.
__global__ void __launch_bounds__(192)
grnn_recur(const float* __restrict__ Gi, const __bf16* __restrict__ WhhB,
           const float* __restrict__ bhh,
           __bf16* __restrict__ hb16, float* __restrict__ hb32,
           float* __restrict__ Yt, float* __restrict__ hlast,
           unsigned* __restrict__ bar) {
    extern __shared__ char smem[];
    const __bf16* Wl = (const __bf16*)(smem + LDS_WL);   // [96][WPAD]
    const __bf16* hT = (const __bf16*)(smem + LDS_HT);   // [16][WPAD]
    float*        ghs = (float*)(smem + LDS_GHS);        // [6][16][16]

    const int cs  = blockIdx.x;     // column slice 0..31 -> j0 = cs*32
    const int mt  = blockIdx.y;     // batch tile 0..3
    const int j0  = cs * 32;
    const int tid = threadIdx.x;
    const int lane = tid & 31;
    const int wave = tid >> 5;      // 0..5
    const int g    = wave >> 1;     // gate: 0=r 1=z 2=n
    const int jj0  = (wave & 1) * 16;
    const int ln   = lane & 15;
    const int lh   = lane >> 4;

    // --- async preload W_hh slice (rows g*H + j0 + jj) into padded LDS ------
    for (int c = tid; c < 96 * 128; c += 192) {          // 128 x 16B chunks/row
        const int row = c >> 7, ch = c & 127;
        const int grow = (row >> 5) * Hsz + j0 + (row & 31);
        async_g2l_b128(LDS_WL + (unsigned)(row * WPAD + ch * 8) * 2u,
                       WhhB + (size_t)grow * Hsz + ch * 8);
    }
    async_wait0();
    __syncthreads();

    const __bf16* wl = Wl + (size_t)(g * 32 + jj0 + ln) * WPAD;
    const __bf16* ha = hT + (size_t)ln * WPAD;

    for (int t = 0; t < Tsz; ++t) {
        const int cur = t & 1, nxt = cur ^ 1;

        // --- async stage this block's 16x1024 h tile into LDS ---------------
        {
            const __bf16* hsrc = hb16 + (size_t)cur * (Bsz * Hsz)
                               + (size_t)(mt * 16) * Hsz;
            for (int c = tid; c < 16 * 128; c += 192) {
                const int row = c >> 7, ch = c & 127;
                async_g2l_b128(LDS_HT + (unsigned)(row * WPAD + ch * 8) * 2u,
                               hsrc + (size_t)row * Hsz + ch * 8);
            }
            async_wait0();
            __syncthreads();
        }

        // gh tile: [16 batch x 16 cols], K = 1024 in 32 WMMA steps (LDS operands)
        v8f acc = {};
        #pragma unroll 4
        for (int ks = 0; ks < 32; ++ks) {
            Frag a, b;
            const int k0 = ks * 32 + lh * 8;
            a.q[0] = *(const u32x4*)(ha + k0);
            a.q[1] = *(const u32x4*)(ha + k0 + 16);
            const int kb = ks * 32 + lh * 8;
            b.q[0] = *(const u32x4*)(wl + kb);
            b.q[1] = *(const u32x4*)(wl + kb + 16);
            acc = __builtin_amdgcn_wmma_f32_16x16x32_bf16(false, a.v, false, b.v,
                                                          (short)0, acc, false, false);
        }

        // stage gh (+ b_hh) to LDS for cross-wave gate combine
        {
            const float bias = bhh[g * Hsz + j0 + jj0 + ln];
            float* gdst = ghs + wave * 256;
            #pragma unroll
            for (int r = 0; r < 8; ++r)
                gdst[(r + 8 * lh) * 16 + ln] = acc[r] + bias;
        }
        __syncthreads();

        // gate combine over 16 rows x 32 cols
        for (int idx = tid; idx < 512; idx += 192) {
            const int rr = idx >> 5;      // batch row in tile
            const int j  = idx & 31;      // column in slice
            const int b  = mt * 16 + rr;
            const size_t girow = ((size_t)b * Tsz + t) * H3;
            const float gi_r = Gi[girow + j0 + j];
            const float gi_z = Gi[girow + Hsz + j0 + j];
            const float gi_n = Gi[girow + 2 * Hsz + j0 + j];
            const int wb = j >> 4, c = j & 15;
            const float gh_r = ghs[(0 + wb) * 256 + rr * 16 + c];
            const float gh_z = ghs[(2 + wb) * 256 + rr * 16 + c];
            const float gh_n = ghs[(4 + wb) * 256 + rr * 16 + c];
            const size_t hoff = (size_t)b * Hsz + j0 + j;
            const float hp = hb32[(size_t)cur * (Bsz * Hsz) + hoff];
            const float r = 1.f / (1.f + __expf(-(gi_r + gh_r)));
            const float z = 1.f / (1.f + __expf(-(gi_z + gh_z)));
            const float n = tanhf(gi_n + r * gh_n);
            const float hn = (1.f - z) * n + z * hp;
            hb32[(size_t)nxt * (Bsz * Hsz) + hoff] = hn;
            hb16[(size_t)nxt * (Bsz * Hsz) + hoff] = (__bf16)hn;
            Yt[((size_t)b * Tsz + t) * Hsz + j0 + j] = hn;
            if (t == Tsz - 1) hlast[hoff] = hn;
        }

        // prefetch next-step Gi rows for this block (global_prefetch_b8)
        if (t + 1 < Tsz && tid < 48) {
            const int pb = mt * 16 + tid / 3, pg = tid % 3;
            __builtin_prefetch(Gi + ((size_t)pb * Tsz + t + 1) * H3 + pg * Hsz + j0, 0, 1);
        }

        // device-wide barrier (monotone counter, one leader per block)
        __threadfence();
        __syncthreads();
        if (tid == 0) {
            atomicAdd(bar, 1u);
            const unsigned need = 128u * (unsigned)(t + 1);
            volatile unsigned* vb = bar;
            while (*vb < need) { __builtin_amdgcn_s_sleep(1); }
        }
        __syncthreads();
        __threadfence();
    }
}

// ------------------------------- launcher ----------------------------------
extern "C" void kernel_launch(void* const* d_in, const int* in_sizes, int n_in,
                              void* d_out, int out_size, void* d_ws, size_t ws_size,
                              hipStream_t stream) {
    (void)in_sizes; (void)n_in; (void)out_size; (void)ws_size;
    const float* Xt   = (const float*)d_in[0];  // [B,T,D]
    const float* Wih  = (const float*)d_in[1];  // [3H,D]
    const float* Whh  = (const float*)d_in[2];  // [3H,H]
    const float* bih  = (const float*)d_in[3];  // [3H]
    const float* bhh  = (const float*)d_in[4];  // [3H]
    const float* h0   = (const float*)d_in[5];  // [B,H]

    float* Yt    = (float*)d_out;                         // [B,T,H]
    float* hlast = Yt + (size_t)Bsz * Tsz * Hsz;          // [B,H]

    char* ws = (char*)d_ws;
    float*    Gi   = (float*)(ws + OFF_GI);
    __bf16*   XtB  = (__bf16*)(ws + OFF_XTB);
    __bf16*   WihB = (__bf16*)(ws + OFF_WIHB);
    __bf16*   WhhB = (__bf16*)(ws + OFF_WHHB);
    float*    hb32 = (float*)(ws + OFF_HB32);
    __bf16*   hb16 = (__bf16*)(ws + OFF_HB16);
    unsigned* bar  = (unsigned*)(ws + OFF_BAR);

    grnn_prep<<<4096, 256, 0, stream>>>(Xt, Wih, Whh, h0, XtB, WihB, WhhB,
                                        hb32, hb16, bar);

    // Gi = Xt @ W_ih^T + b_ih : [32768 x 3072], 128x64 tiles
    grnn_gi_gemm<<<dim3(H3 / 64, (Bsz * Tsz) / 128), 256, 0, stream>>>(XtB, WihB, bih, Gi);

    // persistent recurrence: 32 column slices x 4 batch tiles
    grnn_recur<<<dim3(32, 4), 192, SMEM_TOTAL, stream>>>(Gi, WhhB, bhh, hb16, hb32,
                                                         Yt, hlast, bar);
}
